// GNN_12266426597674
// MI455X (gfx1250) — compile-verified, matrix operand order checked by reading, not run on hardware
//
#include <hip/hip_runtime.h>
#include <hip/hip_bf16.h>
#include <math.h>

// ---------------- problem constants (match reference) ----------------
#define GG   128           // graphs
#define NN   256           // nodes per graph (pre-pool)
#define EMB  128
#define HH   4
#define HC   512           // EMB*H
#define NNODE (GG*NN)      // 32768
#define EDGES (GG*NN*8)    // 262144
#define EPG  (NN*8)        // 2048 edges per graph
#define LLAYERS 3          // layer 3 is dead code in the reference

typedef __attribute__((ext_vector_type(16))) __bf16 v16bf;
typedef __attribute__((ext_vector_type(8)))  float  v8f;
typedef int v4i_ty __attribute__((vector_size(16)));   // matches builtin param type

union FragU { unsigned short s[16]; v16bf v; };

#if defined(__has_builtin)
#if __has_builtin(__builtin_amdgcn_global_load_async_to_lds_b128)
#define HAS_ASYNC_LDS 1
#endif
#if __has_builtin(__builtin_amdgcn_s_wait_asynccnt)
#define HAS_WAIT_ASYNC_BUILTIN 1
#endif
#endif

// global (addrspace 1) int4* and LDS (addrspace 3) int4* casts
#define GAS1(p) ((__attribute__((address_space(1))) v4i_ty*)(p))
#define LAS3(p) ((__attribute__((address_space(3))) v4i_ty*)(p))

__device__ __forceinline__ void wait_async0() {
#if defined(HAS_WAIT_ASYNC_BUILTIN)
  __builtin_amdgcn_s_wait_asynccnt(0);
#else
  asm volatile("s_wait_asynccnt 0" ::: "memory");
#endif
}

__device__ __forceinline__ unsigned short f2bf(float f) {
  unsigned u = __float_as_uint(f);
  unsigned r = (u + 0x7FFFu + ((u >> 16) & 1u)) >> 16;   // RNE
  return (unsigned short)r;
}
__device__ __forceinline__ unsigned fkey(float f) {      // order-preserving key
  unsigned u = __float_as_uint(f);
  return (u & 0x80000000u) ? ~u : (u | 0x80000000u);
}
__device__ __forceinline__ float funkey(unsigned k) {
  unsigned u = (k & 0x80000000u) ? (k & 0x7FFFFFFFu) : ~k;
  return __uint_as_float(u);
}
__device__ __forceinline__ float wave_sum(float s) {
  #pragma unroll
  for (int off = 16; off > 0; off >>= 1) s += __shfl_xor(s, off, 32);
  return s;
}

// ---------------- generic bf16 WMMA GEMM: C = epi(A@W + bias) ----------------
// A:[M,K] f32 row-major, W:[K,N] f32 row-major. 128 threads (4 waves).
// Block tile 128x64; wave w covers rows [w*32, w*32+32) -> 2 strips x 4 col tiles
// = 8 v_wmma per 32-wide K step. Tiles staged to LDS as raw f32 via async DMA
// (GLOBAL_LOAD_ASYNC_TO_LDS_B128) when available; bf16 convert happens in regs.
// epi: 0 = bias, 1 = bias+relu, 2 = bias+relu+batchnorm(g,b,m,v per column)
__global__ __launch_bounds__(128)
void wmma_gemm_kernel(const float* __restrict__ A, const float* __restrict__ W,
                      const float* __restrict__ bias, float* __restrict__ C,
                      int M, int N, int K, int epi,
                      const float* __restrict__ bn_g, const float* __restrict__ bn_b,
                      const float* __restrict__ bn_m, const float* __restrict__ bn_v)
{
  __shared__ float sAf[128 * 32];   // [row][k]   16 KB
  __shared__ float sWf[32 * 64];    // [k][n]      8 KB
  const int bm = blockIdx.x * 128;
  const int bn = blockIdx.y * 64;
  const int tid  = threadIdx.x;
  const int lane = tid & 31;
  const int wave = tid >> 5;               // 0..3 -> 32-row strip pair

  v8f acc[2][4];
  #pragma unroll
  for (int s = 0; s < 2; s++)
    #pragma unroll
    for (int t = 0; t < 4; t++)
      #pragma unroll
      for (int e = 0; e < 8; e++) acc[s][t][e] = 0.0f;

  for (int k0 = 0; k0 < K; k0 += 32) {
#if defined(HAS_ASYNC_LDS)
    // A tile: 128 rows x 32 f32 = 1024 x b128 ; W tile: 32 rows x 64 f32 = 512 x b128
    #pragma unroll
    for (int i = tid; i < 1024; i += 128) {
      int r = i >> 3, seg = i & 7;                    // seg: 4-float chunk
      __builtin_amdgcn_global_load_async_to_lds_b128(
          GAS1(A + (size_t)(bm + r) * K + k0 + seg * 4),
          LAS3(&sAf[r * 32 + seg * 4]), 0, 0);
    }
    #pragma unroll
    for (int i = tid; i < 512; i += 128) {
      int r = i >> 4, seg = i & 15;
      __builtin_amdgcn_global_load_async_to_lds_b128(
          GAS1(W + (size_t)(k0 + r) * N + bn + seg * 4),
          LAS3(&sWf[r * 64 + seg * 4]), 0, 0);
    }
    wait_async0();
#else
    #pragma unroll
    for (int i = tid; i < 128 * 32; i += 128) {
      int r = i >> 5, c = i & 31;
      sAf[r * 32 + c] = A[(size_t)(bm + r) * K + (k0 + c)];
    }
    #pragma unroll
    for (int i = tid; i < 32 * 64; i += 128) {
      int c = i >> 6, n = i & 63;
      sWf[c * 64 + n] = W[(size_t)(k0 + c) * N + (bn + n)];
    }
    if (k0 + 32 < K)
      __builtin_prefetch(&A[(size_t)(bm + tid) * K + (k0 + 32)], 0, 0);
#endif
    __syncthreads();

    const int khalf = (lane >> 4) * 8;     // lanes 16..31 hold the K+8 group
    // B fragments (shared by both row strips)
    FragU bfr[4];
    #pragma unroll
    for (int nt = 0; nt < 4; nt++) {
      const int coln = nt * 16 + (lane & 15);
      #pragma unroll
      for (int e = 0; e < 8; e++) {
        bfr[nt].s[e]     = f2bf(sWf[(khalf + e) * 64 + coln]);
        bfr[nt].s[8 + e] = f2bf(sWf[(16 + khalf + e) * 64 + coln]);
      }
    }
    #pragma unroll
    for (int s = 0; s < 2; s++) {
      const int row = wave * 32 + s * 16 + (lane & 15);
      FragU af;
      #pragma unroll
      for (int e = 0; e < 8; e++) {
        af.s[e]     = f2bf(sAf[row * 32 + khalf + e]);
        af.s[8 + e] = f2bf(sAf[row * 32 + 16 + khalf + e]);
      }
      #pragma unroll
      for (int nt = 0; nt < 4; nt++) {
        acc[s][nt] = __builtin_amdgcn_wmma_f32_16x16x32_bf16(
            false, af.v, false, bfr[nt].v, (short)0, acc[s][nt], false, false);
      }
    }
    __syncthreads();
  }

  // epilogue: C/D layout -> lane col = L%16, VGPR r -> row r + 8*(L>=16)
  #pragma unroll
  for (int nt = 0; nt < 4; nt++) {
    const int coln = bn + nt * 16 + (lane & 15);
    const float bi = bias ? bias[coln] : 0.0f;
    float scale = 1.0f, shift = 0.0f;
    if (epi == 2) {
      scale = __frsqrt_rn(bn_v[coln] + 1e-5f) * bn_g[coln];
      shift = bn_b[coln] - bn_m[coln] * scale;
    }
    #pragma unroll
    for (int s = 0; s < 2; s++) {
      #pragma unroll
      for (int r = 0; r < 8; r++) {
        const int rowm = bm + wave * 32 + s * 16 + r + ((lane >> 4) * 8);
        float val = acc[s][nt][r] + bi;
        if (epi >= 1) val = fmaxf(val, 0.0f);
        if (epi == 2) val = val * scale + shift;
        C[(size_t)rowm * N + coln] = val;
      }
    }
  }
}

// ---------------- fill u32 ----------------
__global__ void fill_u32_kernel(unsigned* __restrict__ p, unsigned val, int n) {
  for (int i = blockIdx.x * blockDim.x + threadIdx.x; i < n; i += gridDim.x * blockDim.x)
    p[i] = val;
}

// ---------------- attention logits: a[e,h] = <q[dst,h,:], k[src,h,:]> / sqrt(EMB) ----------------
__global__ __launch_bounds__(256)
void attn_logits_kernel(const float* __restrict__ q, const float* __restrict__ k,
                        const int* __restrict__ src, const int* __restrict__ dst,
                        float* __restrict__ a)
{
  const int gw   = blockIdx.x * 8 + (threadIdx.x >> 5);
  const int lane = threadIdx.x & 31;
  const int e = gw >> 2, h = gw & 3;
  if (e >= EDGES) return;
  const float4* qv = (const float4*)(q + (size_t)dst[e] * HC + h * EMB);
  const float4* kv = (const float4*)(k + (size_t)src[e] * HC + h * EMB);
  float4 qa = qv[lane], ka = kv[lane];
  float s = qa.x * ka.x + qa.y * ka.y + qa.z * ka.z + qa.w * ka.w;
  s = wave_sum(s);
  if (lane == 0) a[(size_t)e * HH + h] = s * 0.08838834764831845f; // 1/sqrt(128)
}

// ---------------- segment max via ordered-key atomicMax ----------------
__global__ void seg_max_kernel(const float* __restrict__ a, const int* __restrict__ dst,
                               unsigned* __restrict__ amaxk)
{
  int i = blockIdx.x * blockDim.x + threadIdx.x;
  if (i >= EDGES * HH) return;
  int e = i >> 2, h = i & 3;
  atomicMax(&amaxk[(size_t)dst[e] * HH + h], fkey(a[i]));
}

// ---------------- w = exp(a - amax[dst]); den[dst] += w (in-place on a) ----------------
__global__ void exp_den_kernel(float* __restrict__ a, const int* __restrict__ dst,
                               const unsigned* __restrict__ amaxk, float* __restrict__ den)
{
  int i = blockIdx.x * blockDim.x + threadIdx.x;
  if (i >= EDGES * HH) return;
  int e = i >> 2, h = i & 3;
  size_t di = (size_t)dst[e] * HH + h;
  float w = __expf(a[i] - funkey(amaxk[di]));
  a[i] = w;
  atomicAdd(&den[di], w);
}

// ---------------- per-(graph, head, 16-ch chunk) aggregation via LDS atomics ----------------
// S[dst, h*128 + c] += sum_e (w/den) * v[src, h*128 + c]
__global__ __launch_bounds__(256)
void aggregate_kernel(const float* __restrict__ w, const float* __restrict__ den,
                      const float* __restrict__ v, const int* __restrict__ src,
                      const int* __restrict__ dst, float* __restrict__ S)
{
  const int g  = blockIdx.x;
  const int h  = blockIdx.y;
  const int cb = blockIdx.z * 16;
  __shared__ float acc[NN * 16];        // 16 KB
  __shared__ int   s_src[EPG];
  __shared__ short s_dst[EPG];
  __shared__ float s_w[EPG];
  __shared__ float s_den[NN];
  const int tid = threadIdx.x;
  const int e0 = g * EPG;

  for (int i = tid; i < EPG; i += 256) {
    s_src[i] = src[e0 + i];
    s_dst[i] = (short)(dst[e0 + i] - g * NN);
    s_w[i]   = w[(size_t)(e0 + i) * HH + h];
  }
  if (tid < NN) s_den[tid] = den[(size_t)(g * NN + tid) * HH + h];
  for (int i = tid; i < NN * 16; i += 256) acc[i] = 0.0f;
  __syncthreads();

  const int c = tid & 15;
  for (int i = tid >> 4; i < EPG; i += 16) {
    const int dl = s_dst[i];
    const float ww = s_w[i] / (s_den[dl] + 1e-16f);
    const float val = ww * v[(size_t)s_src[i] * HC + h * EMB + cb + c];
    atomicAdd(&acc[dl * 16 + c], val);    // ds_add_f32
  }
  __syncthreads();

  for (int i = tid; i < NN * 16; i += 256) {
    const int n = i >> 4, cc = i & 15;
    const size_t o = (size_t)(g * NN + n) * HC + h * EMB + cb + cc;
    S[o] += acc[i];
  }
}

// ---------------- ||p||^-1 ----------------
__global__ __launch_bounds__(128)
void pnorm_kernel(const float* __restrict__ p, float* __restrict__ invn)
{
  __shared__ float red[128];
  float s = p[threadIdx.x] * p[threadIdx.x];
  red[threadIdx.x] = s;
  __syncthreads();
  for (int st = 64; st > 0; st >>= 1) {
    if (threadIdx.x < st) red[threadIdx.x] += red[threadIdx.x + st];
    __syncthreads();
  }
  if (threadIdx.x == 0) invn[0] = __frsqrt_rn(red[0]);
}

// ---------------- score = tanh(x @ p * invnorm), wave per node ----------------
__global__ __launch_bounds__(256)
void score_kernel(const float* __restrict__ x, const float* __restrict__ p,
                  const float* __restrict__ invn, float* __restrict__ score)
{
  const int node = blockIdx.x * 8 + (threadIdx.x >> 5);
  const int lane = threadIdx.x & 31;
  if (node >= NNODE) return;
  const float4* xv = (const float4*)(x + (size_t)node * EMB);
  const float4* pv = (const float4*)p;
  float4 a = xv[lane], b = pv[lane];
  float s = a.x * b.x + a.y * b.y + a.z * b.z + a.w * b.w;
  s = wave_sum(s);
  if (lane == 0) score[node] = tanhf(s * invn[0]);
}

// ---------------- TopK(ratio=0.5) + gated mean/max readout per graph ----------------
__global__ __launch_bounds__(256)
void pool_rep_kernel(const float* __restrict__ x, const float* __restrict__ score,
                     float* __restrict__ rep)
{
  const int g = blockIdx.x;
  __shared__ float sc[NN];
  __shared__ int   sel[NN];
  const int tid = threadIdx.x;
  sc[tid] = score[g * NN + tid];
  __syncthreads();
  const float mys = sc[tid];
  int rank = 0;
  for (int j = 0; j < NN; j++) {
    float o = sc[j];
    rank += (o > mys) || (o == mys && j < tid);   // stable tie-break like top_k
  }
  sel[tid] = (rank < NN / 2);
  __syncthreads();
  if (tid < EMB) {
    float mean = 0.0f, mx = -3.4e38f;
    for (int n = 0; n < NN; n++) {
      if (sel[n]) {
        float val = x[(size_t)(g * NN + n) * EMB + tid] * sc[n];
        mean += val;
        mx = fmaxf(mx, val);
      }
    }
    rep[g * (2 * EMB) + tid]       = mean * (2.0f / NN);
    rep[g * (2 * EMB) + EMB + tid] = mx;
  }
}

// ---------------- final: out[g] = h2[g,:] @ W3 + b3, wave per graph ----------------
__global__ __launch_bounds__(256)
void final_kernel(const float* __restrict__ h2, const float* __restrict__ W3,
                  const float* __restrict__ b3, float* __restrict__ out)
{
  const int g = blockIdx.x * 8 + (threadIdx.x >> 5);
  const int lane = threadIdx.x & 31;
  if (g >= GG) return;
  const float4* hv = (const float4*)(h2 + (size_t)g * 128);
  const float4* wv = (const float4*)W3;
  float4 a = hv[lane], b = wv[lane];
  float s = a.x * b.x + a.y * b.y + a.z * b.z + a.w * b.w;
  s = wave_sum(s);
  if (lane == 0) out[g] = s + b3[0];
}

// =====================================================================
extern "C" void kernel_launch(void* const* d_in, const int* in_sizes, int n_in,
                              void* d_out, int out_size, void* d_ws, size_t ws_size,
                              hipStream_t stream) {
  (void)in_sizes; (void)n_in; (void)out_size; (void)ws_size;
  const float* x0   = (const float*)d_in[0];
  const float* Wq   = (const float*)d_in[2];
  const float* bq   = (const float*)d_in[3];
  const float* Wk   = (const float*)d_in[4];
  const float* bk   = (const float*)d_in[5];
  const float* Wv   = (const float*)d_in[6];
  const float* bv   = (const float*)d_in[7];
  const float* Ws   = (const float*)d_in[8];
  const float* bs   = (const float*)d_in[9];
  const float* Wt   = (const float*)d_in[10];
  const float* bt   = (const float*)d_in[11];
  const float* bn_g = (const float*)d_in[12];
  const float* bn_b = (const float*)d_in[13];
  const float* bn_m = (const float*)d_in[14];
  const float* bn_v = (const float*)d_in[15];
  const float* poolp= (const float*)d_in[16];
  const float* W1   = (const float*)d_in[17];
  const float* b1   = (const float*)d_in[18];
  const float* W2   = (const float*)d_in[19];
  const float* b2   = (const float*)d_in[20];
  const float* W3   = (const float*)d_in[21];
  const float* b3   = (const float*)d_in[22];
  const int*   ei   = (const int*)d_in[23];
  const int* src = ei;
  const int* dst = ei + EDGES;
  float* out = (float*)d_out;

  // workspace layout (floats)
  float* ws = (float*)d_ws;
  size_t o = 0;
  float* xbuf = ws + o; o += (size_t)NNODE * EMB;     // 4.19M
  float* Qb   = ws + o; o += (size_t)NNODE * HC;      // 16.8M
  float* Kb   = ws + o; o += (size_t)NNODE * HC;
  float* Vb   = ws + o; o += (size_t)NNODE * HC;
  float* Sb   = ws + o; o += (size_t)NNODE * HC;
  float* abuf = ws + o; o += (size_t)EDGES * HH;      // 1.05M
  unsigned* amaxk = (unsigned*)(ws + o); o += (size_t)NNODE * HH;
  float* den  = ws + o; o += (size_t)NNODE * HH;
  float* rep  = ws + o; o += (size_t)GG * 2 * EMB;
  float* h1   = ws + o; o += (size_t)GG * 256;
  float* h2   = ws + o; o += (size_t)GG * 128;
  float* invn = ws + o; o += 1;

  const dim3 blk128(128), blk256(256);
  const dim3 gemm_qkvs(NNODE / 128, HC / 64);         // 256 x 8
  const dim3 gemm_wt(NNODE / 128, EMB / 64);          // 256 x 2
  const int ehTot = EDGES * HH;

  for (int l = 0; l < LLAYERS; l++) {
    const float* xin = (l == 0) ? x0 : xbuf;
    const float* wq = Wq + (size_t)l * EMB * HC, *vq = bq + (size_t)l * HC;
    const float* wk = Wk + (size_t)l * EMB * HC, *vk = bk + (size_t)l * HC;
    const float* wv = Wv + (size_t)l * EMB * HC, *vv = bv + (size_t)l * HC;
    const float* wsk= Ws + (size_t)l * EMB * HC, *vs = bs + (size_t)l * HC;
    const float* wt = Wt + (size_t)l * HC * EMB, *vt = bt + (size_t)l * EMB;

    // Q,K,V,S projections (bf16 WMMA, f32 accumulate)
    wmma_gemm_kernel<<<gemm_qkvs, blk128, 0, stream>>>(xin, wq, vq, Qb, NNODE, HC, EMB, 0, nullptr, nullptr, nullptr, nullptr);
    wmma_gemm_kernel<<<gemm_qkvs, blk128, 0, stream>>>(xin, wk, vk, Kb, NNODE, HC, EMB, 0, nullptr, nullptr, nullptr, nullptr);
    wmma_gemm_kernel<<<gemm_qkvs, blk128, 0, stream>>>(xin, wv, vv, Vb, NNODE, HC, EMB, 0, nullptr, nullptr, nullptr, nullptr);
    wmma_gemm_kernel<<<gemm_qkvs, blk128, 0, stream>>>(xin, wsk, vs, Sb, NNODE, HC, EMB, 0, nullptr, nullptr, nullptr, nullptr);

    // segment softmax state
    fill_u32_kernel<<<256, blk256, 0, stream>>>(amaxk, 0u, NNODE * HH);
    fill_u32_kernel<<<256, blk256, 0, stream>>>((unsigned*)den, 0u, NNODE * HH);

    attn_logits_kernel<<<ehTot / 8, blk256, 0, stream>>>(Qb, Kb, src, dst, abuf);
    seg_max_kernel<<<(ehTot + 255) / 256, blk256, 0, stream>>>(abuf, dst, amaxk);
    exp_den_kernel<<<(ehTot + 255) / 256, blk256, 0, stream>>>(abuf, dst, amaxk, den);

    // weighted aggregation into skip buffer S
    aggregate_kernel<<<dim3(GG, HH, EMB / 16), blk256, 0, stream>>>(abuf, den, Vb, src, dst, Sb);

    // x = BN(relu(S @ Wt + bt))
    wmma_gemm_kernel<<<gemm_wt, blk128, 0, stream>>>(Sb, wt, vt, xbuf, NNODE, EMB, HC, 2,
                                                     bn_g + (size_t)l * EMB, bn_b + (size_t)l * EMB,
                                                     bn_m + (size_t)l * EMB, bn_v + (size_t)l * EMB);
  }

  // TopK pooling + readout (layer-3 conv is dead code in the reference)
  pnorm_kernel<<<1, blk128, 0, stream>>>(poolp, invn);
  score_kernel<<<NNODE / 8, blk256, 0, stream>>>(xbuf, poolp, invn, abuf); // reuse abuf for scores
  pool_rep_kernel<<<GG, blk256, 0, stream>>>(xbuf, abuf, rep);

  // MLP head
  wmma_gemm_kernel<<<dim3(1, 256 / 64), blk128, 0, stream>>>(rep, W1, b1, h1, GG, 256, 2 * EMB, 1, nullptr, nullptr, nullptr, nullptr);
  wmma_gemm_kernel<<<dim3(1, 128 / 64), blk128, 0, stream>>>(h1, W2, b2, h2, GG, 128, 256, 1, nullptr, nullptr, nullptr, nullptr);
  final_kernel<<<GG / 8, blk256, 0, stream>>>(h2, W3, b3, out);
}
